// SelfAttention_39676907883627
// MI455X (gfx1250) — matrix operand order repaired and checked by reading, not compiled, not run on hardware
//
#include <hip/hip_runtime.h>
#include <hip/hip_bf16.h>

#define B_ 4
#define H_ 16
#define S_ 2048
#define D_ 128
#define NTOK (B_ * H_ * S_)   // 131072 token rows
#define NT_  (S_ / 64)        // 32 key tiles

typedef __attribute__((ext_vector_type(16))) _Float16 v16h;
typedef __attribute__((ext_vector_type(8)))  _Float16 v8h;
typedef __attribute__((ext_vector_type(8)))  float    v8f;
typedef __attribute__((ext_vector_type(4)))  unsigned int u32x4;
typedef __attribute__((ext_vector_type(8)))  int  i32x8;
typedef __attribute__((ext_vector_type(4)))  int  i32x4;

// ---- CDNA5 feature probes (device pass only; host pass sees 0) -------------
#if defined(__has_builtin)
#  if __has_builtin(__builtin_amdgcn_tensor_load_to_lds)
#    define HAVE_TDM 1
#  endif
#  if __has_builtin(__builtin_amdgcn_global_load_async_to_lds_b128)
#    define HAVE_ASYNC 1
#  endif
#  if __has_builtin(__builtin_amdgcn_s_wait_asynccnt)
#    define HAVE_WAIT_ASYNC 1
#  endif
#  if __has_builtin(__builtin_amdgcn_s_wait_tensorcnt)
#    define HAVE_WAIT_TENSOR 1
#  endif
#endif
#ifndef HAVE_TDM
#define HAVE_TDM 0
#endif
#ifndef HAVE_ASYNC
#define HAVE_ASYNC 0
#endif
#ifndef HAVE_WAIT_ASYNC
#define HAVE_WAIT_ASYNC 0
#endif
#ifndef HAVE_WAIT_TENSOR
#define HAVE_WAIT_TENSOR 0
#endif

__device__ __forceinline__ v8f vzero8() { v8f z = {}; return z; }

// Low 32 bits of a generic pointer to __shared__ == LDS byte offset (ISA 10.2)
__device__ __forceinline__ unsigned lds_off(const void* p) {
  return (unsigned)(size_t)p;
}

#if HAVE_TDM
// 1-D contiguous TDM copy: ndwords dwords from gaddr -> LDS byte addr.
// D# packing per CDNA5 ISA (group0/group1), type=2.
__device__ __forceinline__ void tdm_load_contig(unsigned lds_byte_addr,
                                                unsigned long long gaddr,
                                                int ndwords) {
  u32x4 g0;
  g0[0] = 1u;                                               // count=1, user D#
  g0[1] = lds_byte_addr;                                    // lds_addr
  g0[2] = (unsigned)gaddr;                                  // global_addr[31:0]
  g0[3] = (unsigned)((gaddr >> 32) & 0x01ffffffull)         // global_addr[56:32]
          | (2u << 30);                                     // type = 2 (image)
  i32x8 g1;
  g1[0] = (int)(2u << 16);                                  // data_size=2 -> 4B
  g1[1] = (int)(((unsigned)ndwords & 0xffffu) << 16);       // tensor_dim0[15:0]
  g1[2] = (int)((((unsigned)ndwords >> 16) & 0xffffu)       // tensor_dim0[31:16]
          | (1u << 16));                                    // tensor_dim1 = 1
  g1[3] = (int)(((unsigned)ndwords & 0xffffu) << 16);       // tile_dim0
  g1[4] = 0;                                                // tile_dim1/2 unused
  g1[5] = ndwords;                                          // dim0 stride lo
  g1[6] = 0;
  g1[7] = 0;
  i32x4 z4 = (i32x4)0;
#if defined(__clang_major__) && __clang_major__ >= 23
  i32x8 z8 = (i32x8)0;
  __builtin_amdgcn_tensor_load_to_lds(g0, g1, z4, z4, z8, 0);
#else
  __builtin_amdgcn_tensor_load_to_lds(g0, g1, z4, z4, 0);
#endif
}
#endif

__device__ __forceinline__ void wait_tensor0() {
#if HAVE_WAIT_TENSOR
  __builtin_amdgcn_s_wait_tensorcnt((short)0);
#else
  asm volatile("s_wait_tensorcnt 0x0" ::: "memory");
#endif
}

__device__ __forceinline__ void wait_async0() {
#if HAVE_WAIT_ASYNC
  __builtin_amdgcn_s_wait_asynccnt(0);
#else
  asm volatile("s_wait_asynccnt 0x0" ::: "memory");
#endif
}

#if HAVE_ASYNC
// Param types per hipcc diagnostic: pointee is a 16-byte int vector;
// source is addrspace(1) ("__device__"), LDS dest addrspace(3).
typedef __attribute__((address_space(1))) i32x4 g_i32x4;
typedef __attribute__((address_space(3))) i32x4 l_i32x4;
// Per-lane async copy of n_vec 16-byte packets, ASYNCcnt-tracked.
__device__ __forceinline__ void async_copy_b128(_Float16* dst_lds,
                                                const _Float16* src,
                                                int tid, int n_vec) {
  for (int i = tid; i < n_vec; i += 128) {
    __builtin_amdgcn_global_load_async_to_lds_b128(
        (g_i32x4*)(unsigned long long)(size_t)(src + (size_t)i * 8),
        (l_i32x4*)(size_t)lds_off(dst_lds + (size_t)i * 8),
        0, 0);
  }
}
#endif

// A-matrix fragment (16x32 f16), contiguous K order (matches B loads):
// lane L holds row m=L&15; elements e=0..15 -> K=kbase+16*half+e.
// Two contiguous v8h loads.
__device__ __forceinline__ v16h wmma_a_frag(const _Float16* __restrict__ row,
                                            int kbase, int half) {
  const v8h lo = *(const v8h*)(row + kbase + 16 * half);
  const v8h hi = *(const v8h*)(row + kbase + 16 * half + 8);
  v16h r;
#pragma unroll
  for (int i = 0; i < 8; ++i) { r[i] = lo[i]; r[i + 8] = hi[i]; }
  return r;
}

// ---------------------------------------------------------------------------
// Kernel 1: q/k/v = x @ W^T + b   (f32 in, f16 out), all GEMMs via WMMA
// grid.x = NTOK/64, block = 128 (4 waves, 16 rows each)
// ---------------------------------------------------------------------------
__global__ __launch_bounds__(128)
void qkv_project_kernel(const float* __restrict__ x1,
                        const float* __restrict__ Wq, const float* __restrict__ bq,
                        const float* __restrict__ Wk, const float* __restrict__ bk,
                        const float* __restrict__ Wv, const float* __restrict__ bv,
                        _Float16* __restrict__ qh, _Float16* __restrict__ kh,
                        _Float16* __restrict__ vh) {
  __shared__ __attribute__((aligned(32))) _Float16 ldsW[D_ * D_]; // 32 KB
  __shared__ __attribute__((aligned(32))) _Float16 ldsX[64 * D_]; // 16 KB

  const int tid  = threadIdx.x;
  const int lane = tid & 31;
  const int wv   = tid >> 5;
  const int half = lane >> 4;
  const int ln16 = lane & 15;
  const long row0 = (long)blockIdx.x * 64;

  // stage 64 token rows of x, f32 -> f16
  for (int i = tid; i < 64 * D_; i += 128)
    ldsX[i] = (_Float16)x1[row0 * D_ + i];
  __syncthreads();

  // per-wave A fragments (16 rows x 128 K, 4 chunks of 32)
  const _Float16* xrow = &ldsX[(wv * 16 + ln16) * D_];
  v16h a[4];
#pragma unroll
  for (int j = 0; j < 4; ++j) a[j] = wmma_a_frag(xrow, 32 * j, half);

  const float* Ws[3] = {Wq, Wk, Wv};
  const float* bs[3] = {bq, bk, bv};
  _Float16*    os[3] = {qh, kh, vh};

#pragma unroll
  for (int m = 0; m < 3; ++m) {
    __syncthreads();
    for (int i = tid; i < D_ * D_; i += 128)
      ldsW[i] = (_Float16)Ws[m][i];
    __syncthreads();

    const float* bias = bs[m];
    _Float16*    dst  = os[m];
#pragma unroll
    for (int nb = 0; nb < 8; ++nb) {
      v8f acc = vzero8();
      // B fragment: lane column e=nb*16+ln16 is a row of W (y = x*W^T),
      // K-dim d contiguous in LDS -> single v16h load
      const _Float16* wrow = &ldsW[(nb * 16 + ln16) * D_];
#pragma unroll
      for (int j = 0; j < 4; ++j) {
        v16h bf = *(const v16h*)(wrow + 32 * j + 16 * half);
        acc = __builtin_amdgcn_wmma_f32_16x16x32_f16(false, a[j], false, bf,
                                                     (short)0, acc, false, false);
      }
      const float bb = bias[nb * 16 + ln16];
#pragma unroll
      for (int r = 0; r < 8; ++r) {
        const int mrow = 8 * half + r;
        dst[(row0 + wv * 16 + mrow) * D_ + nb * 16 + ln16] = (_Float16)(acc[r] + bb);
      }
    }
  }
}

// ---------------------------------------------------------------------------
// Kernel 2: flash attention (no 1/sqrt(d) scaling, per reference).
// grid = (S/64, B*H), block = 128. Each wave owns 16 query rows.
// K tiles double-buffered via TDM (TENSORcnt): tile kt+1 DMA overlaps the
// 32 WMMAs of tile kt.  Q staged by async-to-LDS (ASYNCcnt).
// ---------------------------------------------------------------------------
__global__ __launch_bounds__(128)
void attention_kernel(const _Float16* __restrict__ qh,
                      const _Float16* __restrict__ kh,
                      const _Float16* __restrict__ vh,
                      float* __restrict__ out) {
  __shared__ __attribute__((aligned(32))) _Float16 ldsQ[64 * D_];       // 16 KB
  __shared__ __attribute__((aligned(32))) _Float16 ldsK[2][64 * D_];    // 32 KB [k][d]
  __shared__ __attribute__((aligned(32))) _Float16 ldsVt[2][D_ * 64];   // 32 KB [d][k]
  __shared__ __attribute__((aligned(32))) _Float16 ldsP[4][16 * 64];    // 8 KB

  const int tid  = threadIdx.x;
  const int lane = tid & 31;
  const int wv   = tid >> 5;
  const int half = lane >> 4;
  const int ln16 = lane & 15;
  const int bh   = blockIdx.y;
  const int b    = bh / H_;
  const int h    = bh % H_;
  const long base = (long)bh * S_ * D_;
  const long q0   = (long)blockIdx.x * 64;

  // ---- prologue: stage Q (async) and kick off K tile 0 DMA ----
#if HAVE_ASYNC
  async_copy_b128(ldsQ, qh + base + q0 * D_, tid, 64 * D_ / 8);
#else
  {
    const v8h* src = (const v8h*)(qh + base + q0 * D_);
    v8h* dst = (v8h*)ldsQ;
    for (int i = tid; i < 64 * D_ / 8; i += 128) dst[i] = src[i];
  }
#endif
#if HAVE_TDM
  if (wv == 0)
    tdm_load_contig(lds_off(ldsK[0]),
                    (unsigned long long)(size_t)(kh + base), 64 * D_ / 2);
#elif HAVE_ASYNC
  async_copy_b128(ldsK[0], kh + base, tid, 64 * D_ / 8);
#endif
#if HAVE_ASYNC
  wait_async0();
#endif
  __syncthreads();

  v16h qa[4];
  const _Float16* qrow = &ldsQ[(wv * 16 + ln16) * D_];
#pragma unroll
  for (int j = 0; j < 4; ++j) qa[j] = wmma_a_frag(qrow, 32 * j, half);

  v8f oacc[8];
#pragma unroll
  for (int nb = 0; nb < 8; ++nb) oacc[nb] = vzero8();
  float rmax[8], rsum[8];
#pragma unroll
  for (int r = 0; r < 8; ++r) { rmax[r] = -3.0e38f; rsum[r] = 0.0f; }

  for (int kt = 0; kt < NT_; ++kt) {
    const int  buf   = kt & 1;
    const long kbase = base + (long)kt * 64 * D_;

    // ---- V tile staged transposed: ldsVt[buf][d][k]; overlaps prev PV ----
    for (int i = tid; i < 64 * D_; i += 128) {
      const int krow = i >> 7, d = i & 127;
      ldsVt[buf][d * 64 + krow] = vh[kbase + i];
    }
    if (lane == 0 && kt + 1 < NT_)
      __builtin_prefetch(vh + kbase + 64 * D_, 0, 1);

    // ---- K tile arrival ----
#if HAVE_TDM
    if (wv == 0) wait_tensor0();           // DMA of tile kt complete
#elif HAVE_ASYNC
    wait_async0();
#else
    {
      const v8h* src = (const v8h*)(kh + kbase);
      v8h* dst = (v8h*)ldsK[buf];
      for (int i = tid; i < 64 * D_ / 8; i += 128) dst[i] = src[i];
    }
#endif
    __syncthreads();  // barrier A: K(kt)/V(kt) visible; buf^1 free everywhere

    // ---- kick off DMA of tile kt+1 into the other buffer (overlaps WMMA) ----
#if HAVE_TDM
    if (wv == 0 && kt + 1 < NT_)
      tdm_load_contig(lds_off(ldsK[buf ^ 1]),
                      (unsigned long long)(size_t)(kh + kbase + 64 * D_),
                      64 * D_ / 2);
#elif HAVE_ASYNC
    if (kt + 1 < NT_)
      async_copy_b128(ldsK[buf ^ 1], kh + kbase + 64 * D_, tid, 64 * D_ / 8);
#endif

    // S = Q K^T : 16x64 per wave, 16 WMMAs
    v8f sacc[4];
#pragma unroll
    for (int cb = 0; cb < 4; ++cb) {
      sacc[cb] = vzero8();
      const _Float16* krow = &ldsK[buf][(cb * 16 + ln16) * D_];
#pragma unroll
      for (int j = 0; j < 4; ++j) {
        v16h bf = *(const v16h*)(krow + 32 * j + 16 * half);
        sacc[cb] = __builtin_amdgcn_wmma_f32_16x16x32_f16(false, qa[j], false, bf,
                                                          (short)0, sacc[cb], false, false);
      }
    }

    // online softmax; lane holds rows m=8*half+r at column n=ln16
#pragma unroll
    for (int r = 0; r < 8; ++r) {
      float v = sacc[0][r];
#pragma unroll
      for (int cb = 1; cb < 4; ++cb) v = fmaxf(v, sacc[cb][r]);
#pragma unroll
      for (int off = 1; off < 16; off <<= 1) v = fmaxf(v, __shfl_xor(v, off, 32));
      const float nm    = fmaxf(rmax[r], v);
      const float scale = __expf(rmax[r] - nm);
      rmax[r] = nm;
      float ps = 0.0f;
#pragma unroll
      for (int cb = 0; cb < 4; ++cb) {
        const float p = __expf(sacc[cb][r] - nm);
        ps += p;
        ldsP[wv][(8 * half + r) * 64 + cb * 16 + ln16] = (_Float16)p;
      }
#pragma unroll
      for (int off = 1; off < 16; off <<= 1) ps += __shfl_xor(ps, off, 32);
      rsum[r] = rsum[r] * scale + ps;
#pragma unroll
      for (int nb = 0; nb < 8; ++nb) oacc[nb][r] *= scale;
    }
    __syncthreads();  // P (C-layout) -> LDS -> A-layout reread

    // O += P V : 16 WMMAs
    const _Float16* prow = &ldsP[wv][ln16 * 64];
#pragma unroll
    for (int j2 = 0; j2 < 2; ++j2) {
      v16h pa = wmma_a_frag(prow, 32 * j2, half);
#pragma unroll
      for (int nb = 0; nb < 8; ++nb) {
        const _Float16* vrow = &ldsVt[buf][(nb * 16 + ln16) * 64];
        v16h bf = *(const v16h*)(vrow + 32 * j2 + 16 * half);
        oacc[nb] = __builtin_amdgcn_wmma_f32_16x16x32_f16(false, pa, false, bf,
                                                          (short)0, oacc[nb], false, false);
      }
    }
  }

  // epilogue: out[b, q, h, d] = O / rowsum   (transpose(0,2,1,3) baked in)
#pragma unroll
  for (int r = 0; r < 8; ++r) {
    const float inv  = 1.0f / rsum[r];
    const long  qrow2 = q0 + wv * 16 + 8 * half + r;
#pragma unroll
    for (int nb = 0; nb < 8; ++nb) {
      out[(((long)b * S_ + qrow2) * H_ + h) * D_ + nb * 16 + ln16] = oacc[nb][r] * inv;
    }
  }
}

extern "C" void kernel_launch(void* const* d_in, const int* in_sizes, int n_in,
                              void* d_out, int out_size, void* d_ws, size_t ws_size,
                              hipStream_t stream) {
  const float* x1 = (const float*)d_in[0];
  const float* Wq = (const float*)d_in[1];
  const float* bq = (const float*)d_in[2];
  const float* Wk = (const float*)d_in[3];
  const float* bk = (const float*)d_in[4];
  const float* Wv = (const float*)d_in[5];
  const float* bv = (const float*)d_in[6];

  _Float16* qh = (_Float16*)d_ws;                 // 32 MiB
  _Float16* kh = qh + (size_t)NTOK * D_;          // 32 MiB
  _Float16* vh = kh + (size_t)NTOK * D_;          // 32 MiB

  qkv_project_kernel<<<NTOK / 64, 128, 0, stream>>>(x1, Wq, bq, Wk, bk, Wv, bv,
                                                    qh, kh, vh);
  attention_kernel<<<dim3(S_ / 64, B_ * H_), 128, 0, stream>>>(
      qh, kh, vh, (float*)d_out);
}
